// SyntacticEncoder_85718957293797
// MI455X (gfx1250) — compile-verified, matrix operand order checked by reading, not run on hardware
//
#include <hip/hip_runtime.h>
#include <hip/hip_bf16.h>
#include <cstdint>

// Problem constants (match reference)
#define B_ 8
#define N_ 1024
#define D_ 256
#define H_ 4
#define M_ 3

typedef __attribute__((ext_vector_type(16))) __bf16 v16bf;
typedef __attribute__((ext_vector_type(8)))  float  v8f;
typedef unsigned long long ull;

__device__ __forceinline__ unsigned short f2bf(float f) {
  unsigned int u = __float_as_uint(f);
  unsigned int r = u + 0x7FFFu + ((u >> 16) & 1u);  // round-to-nearest-even
  return (unsigned short)(r >> 16);
}
__device__ __forceinline__ float bf2f(unsigned short h) {
  return __uint_as_float(((unsigned int)h) << 16);
}

union FragU { v16bf v; uint4 q[2]; };

// ---------------------------------------------------------------------------
// Generic tiled bf16 GEMM via v_wmma_f32_16x16x32_bf16 with double-buffered
// LDS fed by GLOBAL_LOAD_ASYNC_TO_LDS_B128 (ASYNCcnt-tracked DMA, no staging
// VGPRs -> no spills, full memory-level parallelism).
//   C[M,Ncols] = A[M,K] * op(B)   (f32 accumulate)
//   BT=false: B given as Bt[Ncols,K] row-major (C = A * Bt^T)  -- torch Linear
//   BT=true : B given as B[K,Ncols] row-major (C = A * B); B tile transposed
//             into LDS via a batched VGPR path (can't transpose in DMA).
//   EPI: 0 none | 1 +bias | 2 +bias,ReLU      OUT_F32: f32 else bf16 output
// Block tile 128x128, 256 threads (8 wave32 = 4x2), wave tile 32x64.
// K staged 64 wide; every M,Ncols is a multiple of 128 and K of 64 here.
// ---------------------------------------------------------------------------
template<bool BT, int EPI, bool OUT_F32>
__global__ __launch_bounds__(256) void gemm_bf16_wmma(
    const unsigned short* __restrict__ Ag, ull sAh, ull sAb, int lda,
    const unsigned short* __restrict__ Bg, ull sBh, ull sBb, int ldb,
    void* __restrict__ Cg, ull sCh, ull sCb, int ldc,
    const float* __restrict__ biasg, ull sBiasH,
    int K, int batchesPerHead)
{
  const int z = blockIdx.z;
  const int h = z / batchesPerHead;
  const int b = z % batchesPerHead;
  const unsigned short* A  = Ag + (ull)h * sAh + (ull)b * sAb;
  const unsigned short* Bm = Bg + (ull)h * sBh + (ull)b * sBb;
  const float* bias = (EPI >= 1) ? (biasg + (ull)h * sBiasH) : nullptr;

  __shared__ unsigned short lA[2][128 * 64];  // [buf][row][k]
  __shared__ unsigned short lB[2][128 * 64];  // [buf][col][k]

  const int tid  = threadIdx.x;
  const int wave = tid >> 5;
  const int lane = tid & 31;
  const int lr   = lane & 15;
  const int hi   = lane >> 4;
  const int waveM = wave >> 1;   // 0..3 -> rows waveM*32
  const int waveN = wave & 1;    // 0..1 -> cols waveN*64
  const int rowBase = blockIdx.y * 128;
  const int colBase = blockIdx.x * 128;

  v8f acc[2][4] = {};

  // --- async DMA staging: global -> LDS, no data VGPRs ------------------
  auto issueAsyncA = [&](int kblk, int buf) {
#pragma unroll
    for (int p = 0; p < 4; ++p) {
      int idx = tid + p * 256;          // 0..1023
      int r   = idx >> 3;               // 128 rows
      int cg  = (idx & 7) * 8;          // 64 k in groups of 8
      unsigned lo = (unsigned)(uintptr_t)&lA[buf][r * 64 + cg];
      const unsigned short* gp = A + (size_t)(rowBase + r) * lda + (size_t)kblk * 64 + cg;
      asm volatile("global_load_async_to_lds_b128 %0, %1, off"
                   :: "v"(lo), "v"(gp) : "memory");
    }
  };
  auto issueAsyncB = [&](int kblk, int buf) {   // !BT only
#pragma unroll
    for (int p = 0; p < 4; ++p) {
      int idx = tid + p * 256;
      int c   = idx >> 3;
      int cg  = (idx & 7) * 8;
      unsigned lo = (unsigned)(uintptr_t)&lB[buf][c * 64 + cg];
      const unsigned short* gp = Bm + (size_t)(colBase + c) * ldb + (size_t)kblk * 64 + cg;
      asm volatile("global_load_async_to_lds_b128 %0, %1, off"
                   :: "v"(lo), "v"(gp) : "memory");
    }
  };
  // BT only: batched load-all-then-store-all transpose into [col][k]
  auto stageBmanual = [&](int kblk, int buf) {
    uint4 t[4];
#pragma unroll
    for (int p = 0; p < 4; ++p) {
      int idx = tid + p * 256;
      int k   = idx >> 4;             // 0..63
      int cg  = (idx & 15) * 8;       // 0..120
      t[p] = *(const uint4*)(Bm + (size_t)(kblk * 64 + k) * ldb + colBase + cg);
    }
#pragma unroll
    for (int p = 0; p < 4; ++p) {
      int idx = tid + p * 256;
      int k   = idx >> 4;
      int cg  = (idx & 15) * 8;
      const unsigned short* sp = (const unsigned short*)&t[p];
#pragma unroll
      for (int j = 0; j < 8; ++j)
        lB[buf][(cg + j) * 64 + k] = sp[j];
    }
  };

  // --- prologue: kick off k-block 0 into buffer 0 -----------------------
  issueAsyncA(0, 0);
  if (!BT) issueAsyncB(0, 0);

  const int nkb = K >> 6;  // K / 64
  for (int kb = 0; kb < nkb; ++kb) {
    const int cur  = kb & 1;
    const bool more = (kb + 1) < nkb;

    // all waves finished reading buf[cur^1] in the previous compute block
    __syncthreads();

    if (more) {
      issueAsyncA(kb + 1, cur ^ 1);
      if (!BT) issueAsyncB(kb + 1, cur ^ 1);
    }
    if (BT) {
      stageBmanual(kb, cur);           // current block's B (transposed)
      if (kb + 2 < nkb)                // gfx1250 global_prefetch_b8
        __builtin_prefetch(Bm + (size_t)((kb + 2) * 64 + (tid >> 4)) * ldb + colBase, 0, 1);
    }

    // retire current buffer's async loads (next block's stay in flight)
    if (more) {
      if (BT) asm volatile("s_wait_asynccnt 0x4" ::: "memory");
      else    asm volatile("s_wait_asynccnt 0x8" ::: "memory");
    } else {
      asm volatile("s_wait_asynccnt 0x0" ::: "memory");
    }
    __syncthreads();                   // cur buffer visible to all waves

    // ---- compute two 32-wide K steps (16 WMMAs) out of buf[cur] ----
#pragma unroll
    for (int ks = 0; ks < 2; ++ks) {
      FragU af[2], bfr[4];
#pragma unroll
      for (int mt = 0; mt < 2; ++mt) {
        int r    = waveM * 32 + mt * 16 + lr;
        int base = r * 64 + ks * 32 + hi * 8;
        af[mt].q[0] = *(const uint4*)(&lA[cur][base]);
        af[mt].q[1] = *(const uint4*)(&lA[cur][base + 16]);
      }
#pragma unroll
      for (int nt = 0; nt < 4; ++nt) {
        int c    = waveN * 64 + nt * 16 + lr;
        int base = c * 64 + ks * 32 + hi * 8;
        bfr[nt].q[0] = *(const uint4*)(&lB[cur][base]);
        bfr[nt].q[1] = *(const uint4*)(&lB[cur][base + 16]);
      }
#pragma unroll
      for (int mt = 0; mt < 2; ++mt)
#pragma unroll
        for (int nt = 0; nt < 4; ++nt)
          acc[mt][nt] = __builtin_amdgcn_wmma_f32_16x16x32_bf16(
              false, af[mt].v, false, bfr[nt].v, (short)0, acc[mt][nt], false, false);
    }
  }

  // ---- epilogue: D frag layout: VGPR r -> row (hi*8 + r), col = lane%16 ----
  float*          Cf = (float*)Cg          + (ull)h * sCh + (ull)b * sCb;
  unsigned short* Ch = (unsigned short*)Cg + (ull)h * sCh + (ull)b * sCb;
#pragma unroll
  for (int mt = 0; mt < 2; ++mt) {
#pragma unroll
    for (int nt = 0; nt < 4; ++nt) {
      int col = colBase + waveN * 64 + nt * 16 + lr;
#pragma unroll
      for (int r = 0; r < 8; ++r) {
        int row = rowBase + waveM * 32 + mt * 16 + hi * 8 + r;
        float v = acc[mt][nt][r];
        if (EPI >= 1) v += bias[col];
        if (EPI == 2) v = fmaxf(v, 0.0f);
        if (OUT_F32) Cf[(size_t)row * ldc + col] = v;
        else         Ch[(size_t)row * ldc + col] = f2bf(v);
      }
    }
  }
}

// ---------------------------------------------------------------------------
__global__ void k_f32_to_bf16(const float* __restrict__ in,
                              unsigned short* __restrict__ out, int n) {
  int i = blockIdx.x * blockDim.x + threadIdx.x;
  if (i < n) out[i] = f2bf(in[i]);
}

// hs[h,b,n,0:D] = h_bf[b,n,:]   (outputs[0] slab of the concat buffer)
__global__ void k_seed_hs(const unsigned short* __restrict__ hbf,
                          unsigned short* __restrict__ hs) {
  int i = blockIdx.x * blockDim.x + threadIdx.x;   // over H*B*N*(D/8)
  const int per = D_ / 8;
  if (i >= H_ * B_ * N_ * per) return;
  int d8 = i % per;
  int t  = i / per;
  int n  = t % N_;
  int t2 = t / N_;
  int b  = t2 % B_;
  int h  = t2 / B_;
  const uint4* src = (const uint4*)(hbf + ((size_t)(b * N_ + n)) * D_ + d8 * 8);
  uint4* dst = (uint4*)(hs + (((size_t)h * B_ + b) * N_ + n) * (4 * D_) + d8 * 8);
  *dst = *src;
}

// In-place row softmax over N_=1024 bf16 elements, scale = 1/sqrt(D)
__global__ __launch_bounds__(256) void k_softmax_row(unsigned short* __restrict__ S) {
  const float scale = 0.0625f;  // 1/sqrt(256)
  unsigned short* p = S + (size_t)blockIdx.x * N_;
  int t = threadIdx.x;
  float vals[4];
  float m = -3.0e38f;
#pragma unroll
  for (int i = 0; i < 4; ++i) {
    vals[i] = bf2f(p[t + i * 256]) * scale;
    m = fmaxf(m, vals[i]);
  }
  __shared__ float red[256];
  red[t] = m; __syncthreads();
  for (int s = 128; s > 0; s >>= 1) { if (t < s) red[t] = fmaxf(red[t], red[t + s]); __syncthreads(); }
  m = red[0]; __syncthreads();
  float sum = 0.0f;
#pragma unroll
  for (int i = 0; i < 4; ++i) { vals[i] = __expf(vals[i] - m); sum += vals[i]; }
  red[t] = sum; __syncthreads();
  for (int s = 128; s > 0; s >>= 1) { if (t < s) red[t] += red[t + s]; __syncthreads(); }
  float inv = 1.0f / red[0];
#pragma unroll
  for (int i = 0; i < 4; ++i) p[t + i * 256] = f2bf(vals[i] * inv);
}

// ---------------------------------------------------------------------------
extern "C" void kernel_launch(void* const* d_in, const int* in_sizes, int n_in,
                              void* d_out, int out_size, void* d_ws, size_t ws_size,
                              hipStream_t stream) {
  (void)in_sizes; (void)n_in; (void)out_size; (void)ws_size;
  const float* nf    = (const float*)d_in[0];
  // d_in[1] = adj_matrix (unused by faithful forward)
  const float* Wq    = (const float*)d_in[2];
  const float* Wk    = (const float*)d_in[3];
  const float* convW = (const float*)d_in[4];
  const float* convB = (const float*)d_in[5];
  const float* projW = (const float*)d_in[6];
  const float* projB = (const float*)d_in[7];
  const float* combW = (const float*)d_in[8];
  const float* combB = (const float*)d_in[9];
  float* out = (float*)d_out;

  // --- workspace carve-out (bf16 intermediates) ---
  char* wsp = (char*)d_ws;
  auto alloc = [&](size_t bytes) -> unsigned short* {
    unsigned short* p = (unsigned short*)wsp;
    wsp += (bytes + 255) & ~(size_t)255;
    return p;
  };
  unsigned short* hbf    = alloc((size_t)B_ * N_ * D_ * 2);          // node feats bf16
  unsigned short* wqbf   = alloc((size_t)H_ * D_ * D_ * 2);
  unsigned short* wkbf   = alloc((size_t)H_ * D_ * D_ * 2);
  unsigned short* cwbf   = alloc((size_t)M_ * H_ * D_ * D_ * 2);
  unsigned short* pwbf   = alloc((size_t)H_ * D_ * 4 * D_ * 2);
  unsigned short* combbf = alloc((size_t)D_ * H_ * D_ * 2);
  unsigned short* Qb     = alloc((size_t)H_ * B_ * N_ * D_ * 2);
  unsigned short* Kb     = alloc((size_t)H_ * B_ * N_ * D_ * 2);
  unsigned short* S      = alloc((size_t)H_ * B_ * N_ * N_ * 2);     // scores -> A
  unsigned short* hs     = alloc((size_t)H_ * B_ * N_ * 4 * D_ * 2); // concat buffer
  unsigned short* agg    = alloc((size_t)H_ * B_ * N_ * D_ * 2);
  unsigned short* hsyn   = alloc((size_t)B_ * N_ * H_ * D_ * 2);     // [B,N,H*D]

  auto cvt = [&](const float* src, unsigned short* dst, int n) {
    k_f32_to_bf16<<<(n + 255) / 256, 256, 0, stream>>>(src, dst, n);
  };
  cvt(nf,    hbf,    B_ * N_ * D_);
  cvt(Wq,    wqbf,   H_ * D_ * D_);
  cvt(Wk,    wkbf,   H_ * D_ * D_);
  cvt(convW, cwbf,   M_ * H_ * D_ * D_);
  cvt(projW, pwbf,   H_ * D_ * 4 * D_);
  cvt(combW, combbf, D_ * H_ * D_);

  const ull sQh = (ull)B_ * N_ * D_, sQb = (ull)N_ * D_;
  const ull sSh = (ull)B_ * N_ * N_, sSb = (ull)N_ * N_;
  const ull sHh = (ull)B_ * N_ * 4 * D_, sHb = (ull)N_ * 4 * D_;

  // Q = h @ Wq^T, K = h @ Wk^T   [per head h, batch b]
  gemm_bf16_wmma<false, 0, false><<<dim3(D_ / 128, N_ / 128, H_ * B_), 256, 0, stream>>>(
      hbf, 0, (ull)N_ * D_, D_, wqbf, (ull)D_ * D_, 0, D_,
      Qb, sQh, sQb, D_, nullptr, 0, D_, B_);
  gemm_bf16_wmma<false, 0, false><<<dim3(D_ / 128, N_ / 128, H_ * B_), 256, 0, stream>>>(
      hbf, 0, (ull)N_ * D_, D_, wkbf, (ull)D_ * D_, 0, D_,
      Kb, sQh, sQb, D_, nullptr, 0, D_, B_);

  // scores = Q @ K^T  (NT: K rows are the columns)
  gemm_bf16_wmma<false, 0, false><<<dim3(N_ / 128, N_ / 128, H_ * B_), 256, 0, stream>>>(
      Qb, sQh, sQb, D_, Kb, sQh, sQb, D_,
      S, sSh, sSb, N_, nullptr, 0, D_, B_);

  // A = softmax(scores * 1/sqrt(D)) in place
  k_softmax_row<<<H_ * B_ * N_, 256, 0, stream>>>(S);

  // outputs[0] slab
  k_seed_hs<<<(H_ * B_ * N_ * (D_ / 8) + 255) / 256, 256, 0, stream>>>(hbf, hs);

  // Densely-connected GCN stack
  for (int m = 0; m < M_; ++m) {
    // agg = A @ h_cur   (NN: h_cur is [K=N, D] row-major slab at col m*D, ldb=4D)
    gemm_bf16_wmma<true, 0, false><<<dim3(D_ / 128, N_ / 128, H_ * B_), 256, 0, stream>>>(
        S, sSh, sSb, N_, hs + m * D_, sHh, sHb, 4 * D_,
        agg, sQh, sQb, D_, nullptr, 0, N_, B_);
    // h_next = relu(agg @ conv_W[m]^T + conv_b[m]) -> slab (m+1)*D
    gemm_bf16_wmma<false, 2, false><<<dim3(D_ / 128, N_ / 128, H_ * B_), 256, 0, stream>>>(
        agg, sQh, sQb, D_, cwbf + (size_t)m * H_ * D_ * D_, (ull)D_ * D_, 0, D_,
        hs + (m + 1) * D_, sHh, sHb, 4 * D_, convB + (size_t)m * H_ * D_, D_, D_, B_);
  }

  // h_syn = D_M @ proj_W^T + proj_b ; write head-concatenated into hsyn[B,N,H*D]
  gemm_bf16_wmma<false, 1, false><<<dim3(D_ / 128, N_ / 128, H_ * B_), 256, 0, stream>>>(
      hs, sHh, sHb, 4 * D_, pwbf, (ull)D_ * 4 * D_, 0, 4 * D_,
      hsyn, /*head col offset*/ (ull)D_, (ull)N_ * H_ * D_, H_ * D_,
      projB, D_, 4 * D_, B_);

  // H_syn = h_out @ comb_W^T + comb_b  -> f32 output
  gemm_bf16_wmma<false, 1, true><<<dim3(D_ / 128, N_ / 128, B_), 256, 0, stream>>>(
      hsyn, 0, (ull)N_ * H_ * D_, H_ * D_, combbf, 0, 0, H_ * D_,
      out, 0, (ull)N_ * D_, D_, combB, 0, H_ * D_, B_);
}
